// TransformerBlock_58901181498006
// MI455X (gfx1250) — compile-verified, matrix operand order checked by reading, not run on hardware
//
#include <hip/hip_runtime.h>

typedef __bf16 bf16_t;
typedef __attribute__((ext_vector_type(16))) __bf16 v16bf;
typedef __attribute__((ext_vector_type(8)))  float  v8f;
typedef int i32x4_ __attribute__((vector_size(16)));

#define T_SEQ  2048
#define C_EMB  4096
#define NHEAD  32
#define HEAD_D 128
#define DFF    16384
#define NQKV   12288

#define EPI_QKV   0
#define EPI_RESID 1
#define EPI_GELU  2

#if defined(__HIP_DEVICE_COMPILE__) && \
    __has_builtin(__builtin_amdgcn_global_load_async_to_lds_b128) && \
    __has_builtin(__builtin_amdgcn_s_wait_asynccnt)
#define ASYNC_LDS 1
#endif

#if defined(ASYNC_LDS)
#define WAIT_ASYNC(n) __builtin_amdgcn_s_wait_asynccnt(n)
#else
#define WAIT_ASYNC(n) ((void)0)
#endif

// 16-byte global -> LDS copy; async (ASYNCcnt-tracked) when available.
static __device__ __forceinline__ void cp16_g2l(bf16_t* ldst, const bf16_t* gsrc) {
#if defined(ASYNC_LDS)
  using gp_t = __attribute__((address_space(1))) i32x4_ *;
  using lp_t = __attribute__((address_space(3))) i32x4_ *;
  __builtin_amdgcn_global_load_async_to_lds_b128((gp_t)(gsrc), (lp_t)(ldst), 0, 0);
#else
  *reinterpret_cast<uint4*>(ldst) = *reinterpret_cast<const uint4*>(gsrc);
#endif
}

static __device__ __forceinline__ v8f vzero8() {
  v8f z = {0.f, 0.f, 0.f, 0.f, 0.f, 0.f, 0.f, 0.f};
  return z;
}

// Load a 16x32 bf16 fragment for v_wmma_f32_16x16x32_bf16:
// lanes 0-15 take K {0..7,16..23}, lanes 16-31 take K {8..15,24..31}.
static __device__ __forceinline__ v16bf ld_frag32(const bf16_t* row_base, int lane_hi) {
  union { uint4 u[2]; v16bf v; } f;
  f.u[0] = *reinterpret_cast<const uint4*>(row_base + lane_hi * 8);
  f.u[1] = *reinterpret_cast<const uint4*>(row_base + 16 + lane_hi * 8);
  return f.v;
}

// ---------------------------------------------------------------------------
// fp32 [K,N] -> bf16 transposed [N,K]  (32x32 tiles through LDS)
// ---------------------------------------------------------------------------
__global__ __launch_bounds__(256) void cvtT_kernel(const float* __restrict__ W,
                                                   bf16_t* __restrict__ Wt,
                                                   int K, int N) {
  __shared__ float t[32][33];
  const int n0 = blockIdx.x * 32, k0 = blockIdx.y * 32;
  const int tx = threadIdx.x & 31, ty = threadIdx.x >> 5;  // ty in 0..7
#pragma unroll
  for (int r = 0; r < 32; r += 8)
    t[ty + r][tx] = W[(size_t)(k0 + ty + r) * N + n0 + tx];
  __syncthreads();
#pragma unroll
  for (int r = 0; r < 32; r += 8)
    Wt[(size_t)(n0 + ty + r) * K + k0 + tx] = (bf16_t)t[tx][ty + r];
}

// ---------------------------------------------------------------------------
// LayerNorm (f32 in) -> bf16 out, one block per row
// ---------------------------------------------------------------------------
__global__ __launch_bounds__(256) void ln_bf16_kernel(const float* __restrict__ x,
                                                      const float* __restrict__ g,
                                                      const float* __restrict__ b,
                                                      bf16_t* __restrict__ out) {
  __shared__ float r1[256];
  __shared__ float r2[256];
  const int row = blockIdx.x;
  const int tid = threadIdx.x;
  float s = 0.f, s2 = 0.f;
  for (int i = tid; i < C_EMB; i += 256) {
    float v = x[(size_t)row * C_EMB + i];
    s += v;
    s2 += v * v;
  }
  r1[tid] = s;
  r2[tid] = s2;
  __syncthreads();
  for (int off = 128; off > 0; off >>= 1) {
    if (tid < off) { r1[tid] += r1[tid + off]; r2[tid] += r2[tid + off]; }
    __syncthreads();
  }
  const float mu   = r1[0] * (1.f / C_EMB);
  const float var  = r2[0] * (1.f / C_EMB) - mu * mu;
  const float rstd = rsqrtf(var + 1e-5f);
  for (int i = tid; i < C_EMB; i += 256) {
    float v = (x[(size_t)row * C_EMB + i] - mu) * rstd * g[i] + b[i];
    out[(size_t)row * C_EMB + i] = (bf16_t)v;
  }
}

// ---------------------------------------------------------------------------
// bf16 WMMA GEMM: C[M,N] = A[M,K] * Wt[N,K]^T (+epilogue)
// 256 thr (8 waves), tile 128x128, BK=32, double-buffered async LDS staging
// ---------------------------------------------------------------------------
__global__ __launch_bounds__(256) void gemm_bf16_kernel(
    const bf16_t* __restrict__ A, const bf16_t* __restrict__ Wt,
    const float* __restrict__ bias, int M, int N, int K, int mode,
    const float* __restrict__ resid, float* __restrict__ outF,
    bf16_t* __restrict__ outB) {
  __shared__ bf16_t lds_a[2][128 * 40];  // [m][k], stride 40 halves (80B, 16B-aligned)
  __shared__ bf16_t lds_b[2][128 * 40];  // [n][k]

  const int tid = threadIdx.x;
  const int lane = tid & 31;
  const int wv = tid >> 5;
  const int lane_lo = lane & 15, lane_hi = lane >> 4;
  const int wave_m = wv >> 2, wave_n = wv & 3;
  const int bn = blockIdx.x, bm = blockIdx.y;

  v8f acc[4][2];
#pragma unroll
  for (int i = 0; i < 4; ++i) { acc[i][0] = vzero8(); acc[i][1] = vzero8(); }

  // stage one 128x32 A tile + 128x32 B tile: 4 x b128 per thread (= 4 async ops/wave)
  auto stage = [&](int kt, int buf) {
#pragma unroll
    for (int j = 0; j < 2; ++j) {
      const int i = tid * 2 + j;      // 0..511
      const int r = i >> 2, kq = i & 3;
      cp16_g2l(&lds_a[buf][r * 40 + kq * 8],
               &A[(size_t)(bm * 128 + r) * K + kt + kq * 8]);
      cp16_g2l(&lds_b[buf][r * 40 + kq * 8],
               &Wt[(size_t)(bn * 128 + r) * K + kt + kq * 8]);
    }
  };

  const int nk = K >> 5;
  stage(0, 0);
  for (int t = 0; t < nk; ++t) {
    const int buf = t & 1;
    if (t + 1 < nk) {
      stage((t + 1) << 5, buf ^ 1);   // overlap next tile's fetch with this compute
      WAIT_ASYNC(4);                  // in-order: tile t's 4 ops have completed
    } else {
      WAIT_ASYNC(0);
    }
    __syncthreads();

    v16bf af[4], bfv[2];
#pragma unroll
    for (int mt = 0; mt < 4; ++mt)
      af[mt] = ld_frag32(&lds_a[buf][(wave_m * 64 + mt * 16 + lane_lo) * 40], lane_hi);
#pragma unroll
    for (int nt = 0; nt < 2; ++nt)
      bfv[nt] = ld_frag32(&lds_b[buf][(wave_n * 32 + nt * 16 + lane_lo) * 40], lane_hi);
#pragma unroll
    for (int mt = 0; mt < 4; ++mt)
#pragma unroll
      for (int nt = 0; nt < 2; ++nt)
        acc[mt][nt] = __builtin_amdgcn_wmma_f32_16x16x32_bf16(
            false, af[mt], false, bfv[nt], (short)0, acc[mt][nt], false, false);
    __syncthreads();  // all waves done with buf before it is restaged
  }

  // epilogue: lane holds col (lane_lo), vgpr j holds row j+8*lane_hi
#pragma unroll
  for (int mt = 0; mt < 4; ++mt) {
#pragma unroll
    for (int nt = 0; nt < 2; ++nt) {
      const int col = bn * 128 + wave_n * 32 + nt * 16 + lane_lo;
      const float bv = bias[col];
#pragma unroll
      for (int j = 0; j < 8; ++j) {
        const int row = bm * 128 + wave_m * 64 + mt * 16 + j + 8 * lane_hi;
        float v = acc[mt][nt][j] + bv;
        if (mode == EPI_QKV) {
          const int s = col >> 12, hh = (col >> 7) & 31, d = col & 127;
          if (s == 0) v *= 0.08838834764831845f;  // 1/sqrt(HEAD_D) folded into Q
          if (s == 2) {
            // V stored transposed: [h][d][T] so attention stages V^T contiguously
            outB[(size_t)2 * NHEAD * T_SEQ * HEAD_D +
                 ((size_t)hh * HEAD_D + d) * T_SEQ + row] = (bf16_t)v;
          } else {
            outB[(((size_t)s * NHEAD + hh) * T_SEQ + row) * HEAD_D + d] = (bf16_t)v;
          }
        } else if (mode == EPI_RESID) {
          outF[(size_t)row * N + col] = resid[(size_t)row * N + col] + v;
        } else {  // EPI_GELU (exact erf-based gelu)
          float gl = 0.5f * v * (1.f + erff(v * 0.70710678118654752f));
          outB[(size_t)row * N + col] = (bf16_t)gl;
        }
      }
    }
  }
}

// ---------------------------------------------------------------------------
// Flash attention: one block = (head, 64-row q tile); 4 waves; WMMA S and PV
// ---------------------------------------------------------------------------
__global__ __launch_bounds__(128) void attn_kernel(const bf16_t* __restrict__ qkv,
                                                   bf16_t* __restrict__ y) {
  __shared__ bf16_t lds_q[64 * 136];    // [q row][d], stride 136 (272B, 16B-aligned)
  __shared__ bf16_t lds_k[64 * 136];    // [key row][d]
  __shared__ bf16_t lds_vt[128 * 72];   // [d][key]
  __shared__ bf16_t lds_p[4 * 16 * 72]; // per-wave P staging [row][key]

  const int h = blockIdx.x;
  const int qt = blockIdx.y;
  const int tid = threadIdx.x;
  const int w = tid >> 5, lane = tid & 31;
  const int lane_lo = lane & 15, lane_hi = lane >> 4;

  const bf16_t* Q   = qkv + ((size_t)(0 * NHEAD + h) * T_SEQ + (size_t)qt * 64) * HEAD_D;
  const bf16_t* Kp  = qkv + ((size_t)(1 * NHEAD + h) * T_SEQ) * HEAD_D;
  const bf16_t* VTp = qkv + (size_t)2 * NHEAD * T_SEQ * HEAD_D + (size_t)h * HEAD_D * T_SEQ;

  // stage Q tile (64x128): 8 x b128 per thread
#pragma unroll
  for (int t = 0; t < 8; ++t) {
    int i = tid + t * 128;
    int r = i >> 4, cq = i & 15;
    cp16_g2l(&lds_q[r * 136 + cq * 8], &Q[(size_t)r * HEAD_D + cq * 8]);
  }

  v8f oacc[8];
  float m_run[8], l_run[8];
#pragma unroll
  for (int i = 0; i < 8; ++i) { oacc[i] = vzero8(); m_run[i] = -1e30f; l_run[i] = 0.f; }

  for (int kt = 0; kt <= qt; ++kt) {
    // stage K tile (64x128) and V^T tile (128x64): straight copies
#pragma unroll
    for (int t = 0; t < 8; ++t) {
      int i = tid + t * 128;
      int r = i >> 4, cq = i & 15;
      cp16_g2l(&lds_k[r * 136 + cq * 8], &Kp[(size_t)(kt * 64 + r) * HEAD_D + cq * 8]);
    }
#pragma unroll
    for (int t = 0; t < 8; ++t) {
      int i = tid + t * 128;
      int d = i >> 3, cq = i & 7;
      cp16_g2l(&lds_vt[d * 72 + cq * 8], &VTp[(size_t)d * T_SEQ + kt * 64 + cq * 8]);
    }
    WAIT_ASYNC(0);
    __syncthreads();

    // S = Q * K^T : wave w owns q rows [w*16, w*16+16), all 64 key cols
    v8f sacc[4];
#pragma unroll
    for (int nt = 0; nt < 4; ++nt) sacc[nt] = vzero8();
#pragma unroll
    for (int nt = 0; nt < 4; ++nt) {
#pragma unroll
      for (int kk = 0; kk < 4; ++kk) {
        v16bf a = ld_frag32(&lds_q[(w * 16 + lane_lo) * 136 + kk * 32], lane_hi);
        v16bf b = ld_frag32(&lds_k[(nt * 16 + lane_lo) * 136 + kk * 32], lane_hi);
        sacc[nt] = __builtin_amdgcn_wmma_f32_16x16x32_bf16(
            false, a, false, b, (short)0, sacc[nt], false, false);
      }
    }

    if (kt == qt) {  // causal mask only on diagonal tile
#pragma unroll
      for (int nt = 0; nt < 4; ++nt)
#pragma unroll
        for (int j = 0; j < 8; ++j)
          if (nt * 16 + lane_lo > w * 16 + j + 8 * lane_hi) sacc[nt][j] = -1e30f;
    }

    // online softmax update, row j of this wave's 16-row band
#pragma unroll
    for (int j = 0; j < 8; ++j) {
      float mt_ = fmaxf(fmaxf(sacc[0][j], sacc[1][j]), fmaxf(sacc[2][j], sacc[3][j]));
#pragma unroll
      for (int off = 1; off < 16; off <<= 1) mt_ = fmaxf(mt_, __shfl_xor(mt_, off, 32));
      const float m_new = fmaxf(m_run[j], mt_);
      const float alpha = __expf(m_run[j] - m_new);
      float rs = 0.f;
#pragma unroll
      for (int nt = 0; nt < 4; ++nt) {
        float pe = __expf(sacc[nt][j] - m_new);
        sacc[nt][j] = pe;
        rs += pe;
      }
#pragma unroll
      for (int off = 1; off < 16; off <<= 1) rs += __shfl_xor(rs, off, 32);
      l_run[j] = l_run[j] * alpha + rs;
      m_run[j] = m_new;
#pragma unroll
      for (int dt = 0; dt < 8; ++dt) oacc[dt][j] *= alpha;
      const int rl = j + 8 * lane_hi;
#pragma unroll
      for (int nt = 0; nt < 4; ++nt)
        lds_p[w * 1152 + rl * 72 + nt * 16 + lane_lo] = (bf16_t)sacc[nt][j];
    }

    // O += P * V (A = P from per-wave LDS, B = V^T)
#pragma unroll
    for (int dt = 0; dt < 8; ++dt) {
#pragma unroll
      for (int kk = 0; kk < 2; ++kk) {
        v16bf a = ld_frag32(&lds_p[w * 1152 + lane_lo * 72 + kk * 32], lane_hi);
        v16bf b = ld_frag32(&lds_vt[(dt * 16 + lane_lo) * 72 + kk * 32], lane_hi);
        oacc[dt] = __builtin_amdgcn_wmma_f32_16x16x32_bf16(
            false, a, false, b, (short)0, oacc[dt], false, false);
      }
    }
    __syncthreads();
  }

  // normalize and store O as bf16 into y[T, C] (head-interleaved layout)
#pragma unroll
  for (int dt = 0; dt < 8; ++dt) {
#pragma unroll
    for (int j = 0; j < 8; ++j) {
      const int row = qt * 64 + w * 16 + j + 8 * lane_hi;
      float o = oacc[dt][j] / l_run[j];
      y[(size_t)row * C_EMB + h * HEAD_D + dt * 16 + lane_lo] = (bf16_t)o;
    }
  }
}

// ---------------------------------------------------------------------------
// Host orchestration
// ---------------------------------------------------------------------------
extern "C" void kernel_launch(void* const* d_in, const int* in_sizes, int n_in,
                              void* d_out, int out_size, void* d_ws, size_t ws_size,
                              hipStream_t stream) {
  (void)in_sizes; (void)n_in; (void)out_size; (void)ws_size;
  const float* x      = (const float*)d_in[0];
  const float* w_attn = (const float*)d_in[1];
  const float* b_attn = (const float*)d_in[2];
  const float* w_proj = (const float*)d_in[3];
  const float* b_proj = (const float*)d_in[4];
  const float* w_fc1  = (const float*)d_in[5];
  const float* b_fc1  = (const float*)d_in[6];
  const float* w_fc2  = (const float*)d_in[7];
  const float* b_fc2  = (const float*)d_in[8];
  const float* ln1g   = (const float*)d_in[9];
  const float* ln1b   = (const float*)d_in[10];
  const float* ln2g   = (const float*)d_in[11];
  const float* ln2b   = (const float*)d_in[12];
  float* out = (float*)d_out;

  char* p = (char*)d_ws;
  auto alloc = [&](size_t elems) {
    bf16_t* r = (bf16_t*)p;
    p += ((elems * 2 + 255) & ~(size_t)255);
    return r;
  };
  bf16_t* wa   = alloc((size_t)C_EMB * NQKV);   // transposed [NQKV, C]
  bf16_t* wp   = alloc((size_t)C_EMB * C_EMB);  // transposed [C, C]
  bf16_t* w1   = alloc((size_t)C_EMB * DFF);    // transposed [DFF, C]
  bf16_t* w2   = alloc((size_t)DFF * C_EMB);    // transposed [C, DFF]
  bf16_t* xn   = alloc((size_t)T_SEQ * C_EMB);
  bf16_t* qkv  = alloc((size_t)3 * NHEAD * T_SEQ * HEAD_D);
  bf16_t* yat  = alloc((size_t)T_SEQ * C_EMB);
  bf16_t* xn2  = alloc((size_t)T_SEQ * C_EMB);
  bf16_t* hmid = alloc((size_t)T_SEQ * DFF);

  // weight fp32 -> bf16 with transpose (so GEMM B staging is contiguous/async)
  cvtT_kernel<<<dim3(NQKV / 32, C_EMB / 32), 256, 0, stream>>>(w_attn, wa, C_EMB, NQKV);
  cvtT_kernel<<<dim3(C_EMB / 32, C_EMB / 32), 256, 0, stream>>>(w_proj, wp, C_EMB, C_EMB);
  cvtT_kernel<<<dim3(DFF / 32, C_EMB / 32), 256, 0, stream>>>(w_fc1, w1, C_EMB, DFF);
  cvtT_kernel<<<dim3(C_EMB / 32, DFF / 32), 256, 0, stream>>>(w_fc2, w2, DFF, C_EMB);

  // xn = LN1(x) in bf16
  ln_bf16_kernel<<<T_SEQ, 256, 0, stream>>>(x, ln1g, ln1b, xn);

  // qkv = xn @ w_attn + b_attn, scattered to [s][h][T][D] bf16 (q scaled, V transposed)
  gemm_bf16_kernel<<<dim3(NQKV / 128, T_SEQ / 128), 256, 0, stream>>>(
      xn, wa, b_attn, T_SEQ, NQKV, C_EMB, EPI_QKV, nullptr, nullptr, qkv);

  // y = softmax(q k^T) v  (causal, flash)
  attn_kernel<<<dim3(NHEAD, T_SEQ / 64), 128, 0, stream>>>(qkv, yat);

  // out = x + y @ w_proj + b_proj   (f32 residual stream)
  gemm_bf16_kernel<<<dim3(C_EMB / 128, T_SEQ / 128), 256, 0, stream>>>(
      yat, wp, b_proj, T_SEQ, C_EMB, C_EMB, EPI_RESID, x, out, nullptr);

  // xn2 = LN2(out)
  ln_bf16_kernel<<<T_SEQ, 256, 0, stream>>>(out, ln2g, ln2b, xn2);

  // hmid = gelu(xn2 @ w_fc1 + b_fc1) in bf16
  gemm_bf16_kernel<<<dim3(DFF / 128, T_SEQ / 128), 256, 0, stream>>>(
      xn2, w1, b_fc1, T_SEQ, DFF, C_EMB, EPI_GELU, nullptr, nullptr, hmid);

  // out = out + hmid @ w_fc2 + b_fc2
  gemm_bf16_kernel<<<dim3(C_EMB / 128, T_SEQ / 128), 256, 0, stream>>>(
      hmid, w2, b_fc2, T_SEQ, C_EMB, DFF, EPI_RESID, out, out, nullptr);
}
